// MultiHeadedAttention_25091198943835
// MI455X (gfx1250) — matrix operand order repaired and checked by reading, not compiled, not run on hardware
//
#include <hip/hip_runtime.h>
#include <hip/hip_bf16.h>

// ---------------------------------------------------------------------------
// MultiHeadedAttention for MI455X (gfx1250).
// All GEMMs via v_wmma_f32_16x16x32_f16. Attention K tiles staged into LDS via
// gfx1250 async-to-LDS copies (ASYNCcnt); V tiles staged via the Tensor Data
// Mover (tensor_load_to_lds, TENSORcnt) using a 2D D# descriptor. Double
// buffered, shared by the 4 waves of each workgroup.
// Pipeline: cvt W->f16 | Q/K/V projections (V stored transposed) |
//           flash attention (online softmax, P via LDS) | output projection.
// ---------------------------------------------------------------------------

typedef _Float16 h16;
typedef __attribute__((ext_vector_type(8)))  _Float16 v8h;
typedef __attribute__((ext_vector_type(16))) _Float16 v16h;
typedef __attribute__((ext_vector_type(8)))  float    v8f;
typedef __attribute__((ext_vector_type(4)))  uint32_t u32x4;
typedef __attribute__((ext_vector_type(4)))  int32_t  i32x4;
typedef __attribute__((ext_vector_type(8)))  int32_t  i32x8;

#define D_MODEL 768
#define NH      12
#define DK      64
#define BATCH   2
#define SEQ     2048
#define MTOT    (BATCH * SEQ)   // 4096 rows

// ---- WMMA wrapper ----------------------------------------------------------
__device__ __forceinline__ v8f wmma_f16(v16h a, v16h b, v8f c) {
  // D = A(16x32 f16) * B(32x16 f16) + C(16x16 f32)
  return __builtin_amdgcn_wmma_f32_16x16x32_f16(
      /*neg_a=*/false, a, /*neg_b=*/false, b,
      /*c_mod=*/(short)0, c, /*reuse_a=*/false, /*reuse_b=*/false);
}

// ---- A-matrix fragment (16x32, per-lane: K chunks {hs*8..+7} and {16+hs*8..+7})
__device__ __forceinline__ v16h a_frag(const h16* rowk0, int hs) {
  const v8h lo = *(const v8h*)(rowk0 + hs * 8);
  const v8h hi = *(const v8h*)(rowk0 + 16 + hs * 8);
  v16h r;
#pragma unroll
  for (int i = 0; i < 8; ++i) { r[i] = lo[i]; r[i + 8] = hi[i]; }
  return r;
}
__device__ __forceinline__ v16h a_frag(const float* rowk0, int hs) {
  const v8f lo = *(const v8f*)(rowk0 + hs * 8);
  const v8f hi = *(const v8f*)(rowk0 + 16 + hs * 8);
  v16h r;
#pragma unroll
  for (int i = 0; i < 8; ++i) { r[i] = (h16)lo[i]; r[i + 8] = (h16)hi[i]; }
  return r;
}

// ---- B-matrix fragment (32x16): lane = column, K chunks {hs*8..+7} and {16+hs*8..+7}
__device__ __forceinline__ v16h b_frag(const h16* colk0, int hs) {
  const v8h lo = *(const v8h*)(colk0 + hs * 8);
  const v8h hi = *(const v8h*)(colk0 + 16 + hs * 8);
  v16h r;
#pragma unroll
  for (int i = 0; i < 8; ++i) { r[i] = lo[i]; r[i + 8] = hi[i]; }
  return r;
}

// ---- gfx1250 async global -> LDS copy (16 bytes), ASYNCcnt-tracked ---------
__device__ __forceinline__ void async_b128(uint32_t lds_off_bytes,
                                           const h16* sbase,
                                           uint32_t voff_bytes) {
  asm volatile("global_load_async_to_lds_b128 %0, %1, %2"
               :
               : "v"(lds_off_bytes), "v"(voff_bytes),
                 "s"((uint64_t)(uintptr_t)sbase)
               : "memory");
}
__device__ __forceinline__ void wait_async0() {
  asm volatile("s_wait_asynccnt 0" ::: "memory");
}
// generic (shared-derived) pointer -> 32-bit LDS byte offset (ISA 10.2: addr[31:0])
__device__ __forceinline__ uint32_t lds_off(const void* p) {
  return (uint32_t)(uintptr_t)p;
}

// ---- gfx1250 Tensor Data Mover: 2D tile (data_size = 2B) global -> LDS -----
// Tile rows are packed contiguously into LDS (X fastest, then Y).
__device__ __forceinline__ void tdm_load_2d(uint32_t lds_addr, const void* gptr,
                                            uint32_t dim0, uint32_t dim1,
                                            uint32_t tile0, uint32_t tile1,
                                            uint32_t stride0) {
  const uint64_t ga = (uint64_t)(uintptr_t)gptr;
  u32x4 g0;
  g0[0] = 1u;                                                  // count=1 (valid)
  g0[1] = lds_addr;                                            // lds_addr
  g0[2] = (uint32_t)ga;                                        // global_addr[31:0]
  g0[3] = ((uint32_t)(ga >> 32) & 0x01FFFFFFu) | 0x80000000u;  // addr[56:32]|type=2
  i32x8 g1;
  g1[0] = 0x00010000;                                          // data_size=2B
  g1[1] = (int)((dim0 & 0xFFFFu) << 16);                       // tensor_dim0 lo
  g1[2] = (int)((dim0 >> 16) | ((dim1 & 0xFFFFu) << 16));      // dim0 hi | dim1 lo
  g1[3] = (int)((dim1 >> 16) | (tile0 << 16));                 // dim1 hi | tile_dim0
  g1[4] = (int)tile1;                                          // tile_dim1 (tile2=0)
  g1[5] = (int)stride0;                                        // dim0_stride[31:0]
  g1[6] = 0;                                                   // stride0 hi, stride1 lo
  g1[7] = 0;
  const i32x4 z4 = {0, 0, 0, 0};
  const i32x8 z8 = {0, 0, 0, 0, 0, 0, 0, 0};
  // 6-arg form (this toolchain): (g0, g1, g2, g3, g4, cpol)
  __builtin_amdgcn_tensor_load_to_lds(g0, g1, z4, z4, z8, 0);
}
__device__ __forceinline__ void wait_tensor0() {
  __builtin_amdgcn_s_wait_tensorcnt((short)0);
}

// ---------------------------------------------------------------------------
// W (f32, [768,768]) -> f16
// ---------------------------------------------------------------------------
__global__ void cvt_w_kernel(const float* __restrict__ W, h16* __restrict__ Wh, int n) {
  int i = blockIdx.x * blockDim.x + threadIdx.x;
  if (i < n) Wh[i] = (h16)W[i];
}

// ---------------------------------------------------------------------------
// Projection: OUT = X @ W^T + b.   Per wave: 16 rows x 64 cols, K-loop of 32.
// MODE 0: f16 out, layout [b, h, S, 64]      (Q, K)
// MODE 1: f16 out, layout [b, h, 64, S]      (V transposed)
// MODE 2: f32 out, layout [M, 768]           (final output)
// Grid: 768 blocks x 128 threads = 3072 waves = 256 M-tiles x 12 N64-tiles.
// ---------------------------------------------------------------------------
template <int MODE, typename TIN>
__global__ __launch_bounds__(128) void proj_kernel(const TIN* __restrict__ X,
                                                   const h16* __restrict__ Wh,
                                                   const float* __restrict__ bias,
                                                   void* __restrict__ out) {
  const int w    = blockIdx.x * 4 + (threadIdx.x >> 5);
  const int lane = threadIdx.x & 31;
  const int hs   = lane >> 4;
  const int l16  = lane & 15;
  const int mtile = w / 12;
  const int ntile = w % 12;
  const int m0 = mtile * 16;
  const int n0 = ntile * 64;

  v8f acc[4] = {};
  const TIN* xrow = X + (size_t)(m0 + l16) * D_MODEL;

  for (int k0 = 0; k0 < D_MODEL; k0 += 32) {
    const v16h aF = a_frag(xrow + k0, hs);
#pragma unroll
    for (int t = 0; t < 4; ++t) {
      const h16* wcol = Wh + (size_t)(n0 + t * 16 + l16) * D_MODEL + k0;
      acc[t] = wmma_f16(aF, b_frag(wcol, hs), acc[t]);
    }
  }

#pragma unroll
  for (int t = 0; t < 4; ++t) {
    const int   n  = n0 + t * 16 + l16;
    const float bv = bias[n];
    if (MODE == 2) {
      float* o = (float*)out;
#pragma unroll
      for (int j = 0; j < 8; ++j) {
        const int m = m0 + j + 8 * hs;
        o[(size_t)m * D_MODEL + n] = acc[t][j] + bv;
      }
    } else if (MODE == 0) {
      h16* o = (h16*)out;
      const int hh = n >> 6, d = n & 63;
#pragma unroll
      for (int j = 0; j < 8; ++j) {
        const int m = m0 + j + 8 * hs;
        const int b = m >> 11, s = m & (SEQ - 1);
        o[(((size_t)(b * NH + hh)) * SEQ + s) * DK + d] = (h16)(acc[t][j] + bv);
      }
    } else {  // MODE 1: V^T [b,h,64,S] -> 8 contiguous halfs per lane
      h16* o = (h16*)out;
      const int hh = n >> 6, d = n & 63;
      const int m = m0 + 8 * hs;
      const int b = m >> 11, s = m & (SEQ - 1);
      v8h pk;
#pragma unroll
      for (int j = 0; j < 8; ++j) pk[j] = (h16)(acc[t][j] + bv);
      *(v8h*)(o + ((size_t)(b * NH + hh) * DK + d) * SEQ + s) = pk;
    }
  }
}

// ---------------------------------------------------------------------------
// Flash attention. One block = 4 waves = one (b,h) x 64 queries (4 q-tiles).
// Per 32-key iteration: K tile (32x64, contiguous) staged via async-to-LDS by
// all threads; V^T tile (64 rows x 32 keys, strided) staged via one TDM
// tensor_load_to_lds issued by wave 0. Double buffered. Each wave then does
// 4 score WMMAs + online softmax + 4 PV WMMAs. Oh: merged heads [M,768] f16.
// Grid: 24 bh * 32 qblocks = 768 blocks x 128 threads.
// ---------------------------------------------------------------------------
struct __align__(128) AttnSmem {
  h16 Ks[2][32 * DK];   // 2 x 4KB  (keys kb..kb+31, row-major [key][d])
  h16 Vs[2][DK * 32];   // 2 x 4KB  (V^T rows   [d][key-kb], packed by TDM)
  h16 P[4][16 * 32];    // 4 x 1KB  per-wave P staging
};

__global__ __launch_bounds__(128) void attn_kernel(const h16* __restrict__ Qh,
                                                   const h16* __restrict__ Kh,
                                                   const h16* __restrict__ VhT,
                                                   h16* __restrict__ Oh) {
  __shared__ AttnSmem sm;

  const int tid  = threadIdx.x;
  const int wv   = tid >> 5;
  const int lane = tid & 31;
  const int hs   = lane >> 4;
  const int l16  = lane & 15;
  const int bh = blockIdx.x >> 5;                  // 24 (b,h) pairs
  const int q0 = ((blockIdx.x & 31) * 4 + wv) * 16;

  const h16* Qb = Qh + (size_t)bh * SEQ * DK;
  const h16* Kb = Kh + (size_t)bh * SEQ * DK;
  const h16* Vb = VhT + (size_t)bh * DK * SEQ;
  h16* lp = sm.P[wv];

  // Q fragments for both K-steps of d (0..31, 32..63)
  const h16* qrow = Qb + (size_t)(q0 + l16) * DK;
  const v16h aQ0 = a_frag(qrow, hs);
  const v16h aQ1 = a_frag(qrow + 32, hs);

  v8f   o[4] = {};
  float mrun[8], lrun[8];
#pragma unroll
  for (int j = 0; j < 8; ++j) { mrun[j] = -1.0e30f; lrun[j] = 0.0f; }
  const float CSC = 0.125f * 1.44269504088896f;  // log2(e) / sqrt(d_k)

  // ---- cooperative stage of one 32-key K/V tile pair -----------------------
  auto stage = [&](int buf, int kb) {
    // K tile: 4KB contiguous in Kh starting at key kb (async-to-LDS path)
    const uint32_t kofs = lds_off(&sm.Ks[buf][0]);
    const uint32_t c0   = (uint32_t)tid * 16u;
    async_b128(kofs + c0, Kb, (uint32_t)(kb * DK * 2) + c0);
    async_b128(kofs + c0 + 2048u, Kb, (uint32_t)(kb * DK * 2) + c0 + 2048u);
    // V tile: 64 rows x 32 halfs, global row stride SEQ (TDM path, wave 0 only)
    if (wv == 0)
      tdm_load_2d(lds_off(&sm.Vs[buf][0]), Vb + kb,
                  /*dim0=*/SEQ, /*dim1=*/DK, /*tile0=*/32, /*tile1=*/DK,
                  /*stride0=*/SEQ);
  };

  stage(0, 0);
  for (int it = 0; it < SEQ / 32; ++it) {
    const int cur = it & 1;
    wait_async0();
    if (wv == 0) wait_tensor0();
    __syncthreads();                               // staged tile visible
    if (it + 1 < SEQ / 32) stage(cur ^ 1, (it + 1) * 32);

    // ---- scores: two 16x16 tiles (16 keys each), K-dim = 64 ----
    v8f sc[2];
#pragma unroll
    for (int t = 0; t < 2; ++t) {
      const h16* kcol = &sm.Ks[cur][(t * 16 + l16) * DK];
      v8f z = {};
      z     = wmma_f16(aQ0, b_frag(kcol, hs), z);
      sc[t] = wmma_f16(aQ1, b_frag(kcol + 32, hs), z);
    }

    // ---- online softmax on raw scores; scale folded into exp2 ----
    float sold[8];
#pragma unroll
    for (int j = 0; j < 8; ++j) {
      float mt = fmaxf(sc[0][j], sc[1][j]);
#pragma unroll
      for (int m = 8; m >= 1; m >>= 1) mt = fmaxf(mt, __shfl_xor(mt, m, 16));
      const float mnew = fmaxf(mrun[j], mt);
      sold[j] = __builtin_amdgcn_exp2f((mrun[j] - mnew) * CSC);
      mrun[j] = mnew;
      sc[0][j] = __builtin_amdgcn_exp2f((sc[0][j] - mnew) * CSC);
      sc[1][j] = __builtin_amdgcn_exp2f((sc[1][j] - mnew) * CSC);
      float rs = sc[0][j] + sc[1][j];
#pragma unroll
      for (int m = 8; m >= 1; m >>= 1) rs += __shfl_xor(rs, m, 16);
      lrun[j] = lrun[j] * sold[j] + rs;
#pragma unroll
      for (int t2 = 0; t2 < 4; ++t2) o[t2][j] *= sold[j];
    }

    // ---- P -> LDS (row-major 16x32), re-read as A fragment ----
#pragma unroll
    for (int t = 0; t < 2; ++t)
#pragma unroll
      for (int j = 0; j < 8; ++j)
        lp[(j + 8 * hs) * 32 + t * 16 + l16] = (h16)sc[t][j];
    asm volatile("s_wait_dscnt 0" ::: "memory");   // cross-lane LDS visibility
    const v16h aP = a_frag(lp + l16 * 32, hs);
    asm volatile("" ::: "memory");

    // ---- O += P V : 4 n-tiles of d, K-dim = 32 keys ----
#pragma unroll
    for (int t2 = 0; t2 < 4; ++t2) {
      const h16* vcol = &sm.Vs[cur][(t2 * 16 + l16) * 32];
      o[t2] = wmma_f16(aP, b_frag(vcol, hs), o[t2]);
    }
    __syncthreads();                               // all readers done with cur
  }

  // ---- normalize + store merged-head f16 output ----
  const int b = bh / NH, hh = bh % NH;
#pragma unroll
  for (int t2 = 0; t2 < 4; ++t2) {
    const int d = t2 * 16 + l16;
#pragma unroll
    for (int j = 0; j < 8; ++j) {
      const int s = q0 + j + 8 * hs;
      Oh[((size_t)(b * SEQ + s)) * D_MODEL + hh * DK + d] = (h16)(o[t2][j] / lrun[j]);
    }
  }
}

// ---------------------------------------------------------------------------
extern "C" void kernel_launch(void* const* d_in, const int* in_sizes, int n_in,
                              void* d_out, int out_size, void* d_ws, size_t ws_size,
                              hipStream_t stream) {
  const float* q = (const float*)d_in[0];
  const float* k = (const float*)d_in[1];
  const float* v = (const float*)d_in[2];
  const float* W = (const float*)d_in[3];
  const float* b = (const float*)d_in[4];

  // f16 workspace: W | Q | K | V^T | O  (~26.3 MB)
  h16* Wh  = (h16*)d_ws;
  h16* Qh  = Wh + (size_t)D_MODEL * D_MODEL;
  h16* Kh  = Qh + (size_t)MTOT * D_MODEL;
  h16* VhT = Kh + (size_t)MTOT * D_MODEL;
  h16* Oh  = VhT + (size_t)MTOT * D_MODEL;

  cvt_w_kernel<<<(D_MODEL * D_MODEL) / 256, 256, 0, stream>>>(W, Wh, D_MODEL * D_MODEL);

  proj_kernel<0, float><<<768, 128, 0, stream>>>(q, Wh, b, Qh);
  proj_kernel<0, float><<<768, 128, 0, stream>>>(k, Wh, b, Kh);
  proj_kernel<1, float><<<768, 128, 0, stream>>>(v, Wh, b, VhT);

  attn_kernel<<<768, 128, 0, stream>>>(Qh, Kh, VhT, Oh);

  proj_kernel<2, h16><<<768, 128, 0, stream>>>(Oh, Wh, b, (float*)d_out);
}
